// AnchorStripeAttention_3865470566918
// MI455X (gfx1250) — matrix operand order repaired and checked by reading, not compiled, not run on hardware
//
#include <hip/hip_runtime.h>
#include <math.h>

// ---------------------------------------------------------------------------
// Anchor Stripe Attention for MI455X (gfx1250), wave32 + WMMA f16 16x16x32.
// One wave per (window, head); 6 waves (192 threads) per block = 1 window.
// Both attention matrices are computed TRANSPOSED so the softmax axis lies on
// C-layout rows: reductions become in-lane + one xor16 shuffle, bias/mask load
// as float4 in native layout, prob/e-value stores pack as ds_store_b128, and
// outputs store as global_store_b128. Norm factors via wmma-with-ones.
// ---------------------------------------------------------------------------

typedef __attribute__((ext_vector_type(16))) _Float16 v16h;
typedef __attribute__((ext_vector_type(2)))  __fp16   v2fp;   // cvt_pkrtz result
typedef __attribute__((ext_vector_type(8)))  float    v8f;

#define NHEAD 6
#define DHEAD 32
#define NWIN  2048
#define LOG100 4.605170185988091f

// per-wave LDS layout in halfs (all row strides keep 16B alignment)
#define KN_OFF   0          // k raw f16 : 64 rows x stride 40
#define QN_OFF   2560       // q raw f16 : 64 x 40
#define AN_OFF   5120       // anc raw   : 16 x 40
#define VT_OFF   5760       // v dim-major: 32 rows x stride 72 (64 tokens)
#define A1_OFF   0          // attn1^T e-vals as [anchor][key]: 16 x 72 (ovl kn)
#define A2_OFF   0          // attn2^T e-vals as [q][anchor]: 64 x 40 (ovl kn/at1)
#define X1_OFF   5760       // x1 dim-major [dim][anchor]: 32 x 40 (ovl vt)
#define PER_WAVE 8064       // halfs per wave (16128 B); 6 waves = 94.5KB

union V16U { v16h v; uint4 q[2]; };
union H4U  { v2fp h2[2]; uint2 u; };
union H8U  { v2fp h2[4]; uint4 u; };
union F8U  { float4 v[2]; float f[8]; };

// A operand (16xK tile at c0): lane = row (lane&15);
// halves {c0+sel*8..+8} and {c0+16+sel*8..+8}, sel = lane>>4.
__device__ __forceinline__ v16h load_A(const _Float16* base, int row0, int stride,
                                       int c0, int lane) {
    const int r   = row0 + (lane & 15);
    const int sel = lane >> 4;
    const _Float16* p = base + r * stride + c0;
    V16U a;
    a.q[0] = *(const uint4*)(p + sel * 8);
    a.q[1] = *(const uint4*)(p + 16 + sel * 8);
    return a.v;
}

// B operand (32x16 tile): lane = column (lane&15) stored as a row of `base`
// (N-major storage); halves {c0+sel*8..+8} and {c0+16+sel*8..+8}, sel = lane>>4.
__device__ __forceinline__ v16h load_B(const _Float16* base, int col0, int stride,
                                       int c0, int lane) {
    const int n   = col0 + (lane & 15);
    const int sel = lane >> 4;
    const _Float16* p = base + n * stride + c0;
    V16U b;
    b.q[0] = *(const uint4*)(p + sel * 8);
    b.q[1] = *(const uint4*)(p + 16 + sel * 8);
    return b.v;
}

__device__ __forceinline__ v8f wmma_f16(v16h a, v16h b, v8f c) {
    return __builtin_amdgcn_wmma_f32_16x16x32_f16(false, a, false, b,
                                                  (short)0, c, false, false);
}

// 1/max(sqrt(ss),1e-12) via v_rsq_f32 (rsq(0)=inf -> clamp to 1e12).
__device__ __forceinline__ float rsq_c(float ss) {
    return fminf(__builtin_amdgcn_rsqf(ss), 1e12f);
}

__device__ __forceinline__ void load_f8(const float* p, float f[8]) {
    F8U u;
    u.v[0] = *(const float4*)(p);
    u.v[1] = *(const float4*)(p + 4);
#pragma unroll
    for (int i = 0; i < 8; ++i) f[i] = u.f[i];
}

// ---------------------------------------------------------------------------
// Kernel 1: CPB-MLP tables. bt[which][121][6] = relu(table @ W1 + b1) @ W2.
// ---------------------------------------------------------------------------
__global__ void asa_cpb_kernel(const float* __restrict__ table,
                               const float* __restrict__ w1a, const float* __restrict__ b1a,
                               const float* __restrict__ w2a,
                               const float* __restrict__ w1b, const float* __restrict__ b1b,
                               const float* __restrict__ w2b,
                               float* __restrict__ bt) {
    const int t     = blockIdx.x;
    const int which = blockIdx.y;
    const float* w1 = which ? w1b : w1a;
    const float* b1 = which ? b1b : b1a;
    const float* w2 = which ? w2b : w2a;
    const float x = table[t * 2 + 0];
    const float y = table[t * 2 + 1];
    const int j = threadIdx.x;

    float hsum[NHEAD];
#pragma unroll
    for (int hh = 0; hh < NHEAD; ++hh) hsum[hh] = 0.f;
    for (int jj = j; jj < 512; jj += 256) {
        float hv = fmaxf(x * w1[jj] + y * w1[512 + jj] + b1[jj], 0.f);
#pragma unroll
        for (int hh = 0; hh < NHEAD; ++hh) hsum[hh] += hv * w2[jj * NHEAD + hh];
    }
    __shared__ float red[256];
    for (int hh = 0; hh < NHEAD; ++hh) {
        red[j] = hsum[hh];
        __syncthreads();
        for (int off = 128; off; off >>= 1) {
            if (j < off) red[j] += red[j + off];
            __syncthreads();
        }
        if (j == 0) bt[which * 726 + t * NHEAD + hh] = red[0];
        __syncthreads();
    }
}

// ---------------------------------------------------------------------------
// Kernel 2: gather + 16*sigmoid into dense bias tensors (native layouts:
// bias1 = [h][anchor][key], bias2 = [h][q][anchor]).
// ---------------------------------------------------------------------------
__global__ void asa_gather_kernel(const int* __restrict__ idx1, const int* __restrict__ idx2,
                                  const float* __restrict__ bt,
                                  float* __restrict__ bias1, float* __restrict__ bias2) {
    const int i = blockIdx.x * 256 + threadIdx.x;
    if (i < NHEAD * 1024) {
        const int h = i / 1024, mn = i % 1024;
        {   const int t = idx1[mn];
            bias1[i] = 16.f / (1.f + __expf(-bt[t * NHEAD + h])); }
        {   const int t = idx2[mn];
            bias2[i] = 16.f / (1.f + __expf(-bt[726 + t * NHEAD + h])); }
    }
}

// ---------------------------------------------------------------------------
// Main kernel: one block per window, one wave per head.
// ---------------------------------------------------------------------------
extern __shared__ __align__(16) _Float16 asa_smem[];

__global__ __launch_bounds__(NHEAD * 32) void asa_main_kernel(
    const float* __restrict__ qkv, const float* __restrict__ anchor,
    const float* __restrict__ mask1, const float* __restrict__ mask2,
    const float* __restrict__ ls1, const float* __restrict__ ls2,
    const float* __restrict__ bias1, const float* __restrict__ bias2,
    float* __restrict__ out) {
    const int lane = threadIdx.x & 31;
    const int h    = threadIdx.x >> 5;
    const int w    = blockIdx.x;
    const int b    = w >> 10;
    const int wi   = w & 1023;
    const int wy   = wi >> 5, wx = wi & 31;
    const int nlo  = lane & 15, hi = lane >> 4;

    _Float16* base = asa_smem + h * PER_WAVE;
    _Float16* kn   = base + KN_OFF;
    _Float16* qn   = base + QN_OFF;
    _Float16* anc  = base + AN_OFF;
    _Float16* vt   = base + VT_OFF;
    _Float16* at1  = base + A1_OFF;   // overlays kn (dead after stage-1 GEMMs)
    _Float16* at2q = base + A2_OFF;   // overlays kn/at1 (dead after x1 GEMMs)
    _Float16* x1dm = base + X1_OFF;   // overlays vt (dead after x1 GEMMs)

    const float scale1 = __expf(fminf(ls1[h], LOG100));
    const float scale2 = __expf(fminf(ls2[h], LOG100));

    const size_t winpix = (size_t)b * 65536 + (size_t)wy * 2048 + (size_t)wx * 8;
    const size_t ancpix = (size_t)b * 16384 + (size_t)wy * 512  + (size_t)wx * 4;

    // ---- Load q,k,v raw f16 (no pre-normalization). float4 loads, pkrtz cvt.
    const int tsub = lane >> 3, d0 = (lane & 7) * 4;
#pragma unroll 4
    for (int tb = 0; tb < 16; ++tb) {
        const int t = tb * 4 + tsub;
        const size_t pix = winpix + (size_t)(t >> 3) * 256 + (t & 7);
        const float4* p = (const float4*)(qkv + pix * 576 + h * DHEAD + d0);
        float4 qv = p[0];
        float4 kv = p[48];    // +192 floats
        float4 vv = p[96];    // +384 floats
        H4U q4, k4;
        q4.h2[0] = __builtin_amdgcn_cvt_pkrtz(qv.x, qv.y);
        q4.h2[1] = __builtin_amdgcn_cvt_pkrtz(qv.z, qv.w);
        k4.h2[0] = __builtin_amdgcn_cvt_pkrtz(kv.x, kv.y);
        k4.h2[1] = __builtin_amdgcn_cvt_pkrtz(kv.z, kv.w);
        *(uint2*)(qn + t * 40 + d0) = q4.u;
        *(uint2*)(kn + t * 40 + d0) = k4.u;
        vt[(d0 + 0) * 72 + t] = (_Float16)vv.x;
        vt[(d0 + 1) * 72 + t] = (_Float16)vv.y;
        vt[(d0 + 2) * 72 + t] = (_Float16)vv.z;
        vt[(d0 + 3) * 72 + t] = (_Float16)vv.w;
    }
#pragma unroll
    for (int ab = 0; ab < 4; ++ab) {
        const int a = ab * 4 + tsub;
        const size_t pix = ancpix + (size_t)(a >> 2) * 128 + (a & 3);
        const float4* p = (const float4*)(anchor + pix * 192 + h * DHEAD + d0);
        float4 av = p[0];
        H4U a4;
        a4.h2[0] = __builtin_amdgcn_cvt_pkrtz(av.x, av.y);
        a4.h2[1] = __builtin_amdgcn_cvt_pkrtz(av.z, av.w);
        *(uint2*)(anc + a * 40 + d0) = a4.u;
    }

    v16h onesv;
#pragma unroll
    for (int i = 0; i < 16; ++i) onesv[i] = (_Float16)1.0f;
    const v8f zz = {};

    // ---- Anchor norm factors via WMMA broadcast.
    const v16h aA   = load_A(anc, 0, 40, 0, lane);   // anchors as rows
    const v16h bAnc = load_B(anc, 0, 40, 0, lane);   // anchors as cols
    v8f ssac = wmma_f16(onesv, bAnc * bAnc, zz);     // per-lane (col) anchor ss
    v8f ssar = wmma_f16(aA * aA, onesv, zz);         // per-g (row) anchor ss
    const float invAc = rsq_c(ssac[0]);
    float invAr[8];
#pragma unroll
    for (int g = 0; g < 8; ++g) invAr[g] = rsq_c(ssar[g]);

    // ---- Stage 1 (transposed): attn1^T(64 keys x 16 anchors) = k @ anc^T.
    v8f acc[4], ssk[4];
#pragma unroll
    for (int rt = 0; rt < 4; ++rt) {
        const v16h aK = load_A(kn, rt * 16, 40, 0, lane);
        acc[rt] = wmma_f16(aK, bAnc, zz);
        ssk[rt] = wmma_f16(aK * aK, onesv, zz);      // per-key (row) sumsq
    }
    // logits: lane owns anchor column nlo, keys rt*16 + 8*hi + g (contiguous).
    float tl[4][8];
    const float s1 = scale1 * invAc;
#pragma unroll
    for (int rt = 0; rt < 4; ++rt) {
        float bb[8], mm[8];
        load_f8(bias1 + h * 1024 + nlo * 64 + rt * 16 + 8 * hi, bb);
        load_f8(mask1 + (size_t)wi * 1024 + nlo * 64 + rt * 16 + 8 * hi, mm);
#pragma unroll
        for (int g = 0; g < 8; ++g)
            tl[rt][g] = acc[rt][g] * (s1 * rsq_c(ssk[rt][g])) + bb[g] + mm[g];
    }
    // softmax over keys = rows: in-lane max/sum + one xor16 shuffle.
    float mx = tl[0][0];
#pragma unroll
    for (int rt = 0; rt < 4; ++rt)
#pragma unroll
        for (int g = 0; g < 8; ++g) mx = fmaxf(mx, tl[rt][g]);
    mx = fmaxf(mx, __shfl_xor(mx, 16, 32));
    float s = 0.f;
#pragma unroll
    for (int rt = 0; rt < 4; ++rt) {
        float e[8];
#pragma unroll
        for (int g = 0; g < 8; ++g) { e[g] = __expf(tl[rt][g] - mx); s += e[g]; }
        H8U pk;
#pragma unroll
        for (int g = 0; g < 8; g += 2)
            pk.h2[g >> 1] = __builtin_amdgcn_cvt_pkrtz(e[g], e[g + 1]);
        // at1[anchor][key]: 8 consecutive halves -> one ds_store_b128
        *(uint4*)(at1 + nlo * 72 + rt * 16 + 8 * hi) = pk.u;
    }
    s += __shfl_xor(s, 16, 32);
    const float invr1 = __builtin_amdgcn_rcpf(s);    // per-lane (anchor) 1/rowsum

    // ---- x1^T(32 dims x 16 anchors) = v^T @ E1; scale cols by invr1.
    const v16h bE0 = load_B(at1, 0, 72, 0, lane);
    const v16h bE1 = load_B(at1, 0, 72, 32, lane);
    v8f xc[2];
#pragma unroll
    for (int dt = 0; dt < 2; ++dt) {
        v8f c = wmma_f16(load_A(vt, dt * 16, 72, 0, lane), bE0, zz);
        c = wmma_f16(load_A(vt, dt * 16, 72, 32, lane), bE1, c);
        xc[dt] = c;
    }
    // vt dead -> store x1 dim-major [dim][anchor] (A operand of final GEMM).
#pragma unroll
    for (int dt = 0; dt < 2; ++dt)
#pragma unroll
        for (int g = 0; g < 8; ++g)
            x1dm[(dt * 16 + g + 8 * hi) * 40 + nlo] = (_Float16)(xc[dt][g] * invr1);
    // zero K-padding: x1dm cols 16..31 (anchors) for all 32 dim rows.
    *(uint4*)(x1dm + lane * 40 + 16) = make_uint4(0u, 0u, 0u, 0u);
    *(uint4*)(x1dm + lane * 40 + 24) = make_uint4(0u, 0u, 0u, 0u);
    // at1 dead -> at2q[q][anchor] K-padding (cols 16..31) for all 64 q rows.
#pragma unroll
    for (int i = 0; i < 2; ++i) {
        *(uint4*)(at2q + (lane * 2 + i) * 40 + 16) = make_uint4(0u, 0u, 0u, 0u);
        *(uint4*)(at2q + (lane * 2 + i) * 40 + 24) = make_uint4(0u, 0u, 0u, 0u);
    }

    // ---- Stage 2 (transposed): attn2^T(16 anchors x 64 q) = anc @ q^T.
    float invr2[4];
#pragma unroll
    for (int jt = 0; jt < 4; ++jt) {
        const v16h bQ = load_B(qn, jt * 16, 40, 0, lane);
        v8f c2  = wmma_f16(aA, bQ, zz);
        v8f ssq = wmma_f16(onesv, bQ * bQ, zz);      // per-lane (q col) sumsq
        const float s2 = scale2 * rsq_c(ssq[0]);
        float bb[8], mm[8];
        // lane owns q column jt*16+nlo; anchors 8*hi+g contiguous in native layout
        load_f8(bias2 + h * 1024 + (jt * 16 + nlo) * 16 + 8 * hi, bb);
        load_f8(mask2 + (size_t)wi * 1024 + (jt * 16 + nlo) * 16 + 8 * hi, mm);
        float t2[8];
#pragma unroll
        for (int g = 0; g < 8; ++g)
            t2[g] = c2[g] * (s2 * invAr[g]) + bb[g] + mm[g];
        float m2 = t2[0];
#pragma unroll
        for (int g = 1; g < 8; ++g) m2 = fmaxf(m2, t2[g]);
        m2 = fmaxf(m2, __shfl_xor(m2, 16, 32));
        float s2s = 0.f;
        float e[8];
#pragma unroll
        for (int g = 0; g < 8; ++g) { e[g] = __expf(t2[g] - m2); s2s += e[g]; }
        H8U pk;
#pragma unroll
        for (int g = 0; g < 8; g += 2)
            pk.h2[g >> 1] = __builtin_amdgcn_cvt_pkrtz(e[g], e[g + 1]);
        *(uint4*)(at2q + (jt * 16 + nlo) * 40 + 8 * hi) = pk.u;
        s2s += __shfl_xor(s2s, 16, 32);
        invr2[jt] = __builtin_amdgcn_rcpf(s2s);      // per-lane (q) 1/rowsum
    }

    // ---- out^T(32 dims x 64 q) = x1^T @ E2; lane owns one q token's 8 dims
    // (contiguous) -> 2x global_store_b128 per tile.
#pragma unroll
    for (int dt = 0; dt < 2; ++dt) {
        const v16h aX = load_A(x1dm, dt * 16, 40, 0, lane);
#pragma unroll
        for (int jt = 0; jt < 4; ++jt) {
            v8f o = wmma_f16(aX, load_B(at2q, jt * 16, 40, 0, lane), zz);
            const float ir = invr2[jt];
            const int q = jt * 16 + nlo;
            const size_t pix = winpix + (size_t)(q >> 3) * 256 + (q & 7);
            float* op = out + pix * 192 + h * DHEAD + dt * 16 + 8 * hi;
            *(float4*)(op)     = make_float4(o[0] * ir, o[1] * ir, o[2] * ir, o[3] * ir);
            *(float4*)(op + 4) = make_float4(o[4] * ir, o[5] * ir, o[6] * ir, o[7] * ir);
        }
    }
}

// ---------------------------------------------------------------------------
extern "C" void kernel_launch(void* const* d_in, const int* in_sizes, int n_in,
                              void* d_out, int out_size, void* d_ws, size_t ws_size,
                              hipStream_t stream) {
    const float* qkv    = (const float*)d_in[0];
    const float* anchor = (const float*)d_in[1];
    const float* table  = (const float*)d_in[2];
    const int*   idx1   = (const int*)d_in[3];
    const int*   idx2   = (const int*)d_in[4];
    const float* mask1  = (const float*)d_in[5];
    const float* mask2  = (const float*)d_in[6];
    const float* ls1    = (const float*)d_in[7];
    const float* ls2    = (const float*)d_in[8];
    const float* w1a    = (const float*)d_in[9];
    const float* b1a    = (const float*)d_in[10];
    const float* w2a    = (const float*)d_in[11];
    const float* w1b    = (const float*)d_in[12];
    const float* b1b    = (const float*)d_in[13];
    const float* w2b    = (const float*)d_in[14];
    float* out = (float*)d_out;

    float* ws    = (float*)d_ws;
    float* bt    = ws;            // [2][121][6]
    float* bias1 = ws + 1452;     // [6][16][64]
    float* bias2 = ws + 7596;     // [6][64][16]

    dim3 g1(121, 2);
    asa_cpb_kernel<<<g1, 256, 0, stream>>>(table, w1a, b1a, w2a, w1b, b1b, w2b, bt);
    asa_gather_kernel<<<24, 256, 0, stream>>>(idx1, idx2, bt, bias1, bias2);

    const size_t smem = (size_t)NHEAD * PER_WAVE * sizeof(_Float16); // 96768 B
    asa_main_kernel<<<NWIN, NHEAD * 32, smem, stream>>>(
        qkv, anchor, mask1, mask2, ls1, ls2, bias1, bias2, out);
}
